// FovealAttentionBlock_11845519802847
// MI455X (gfx1250) — compile-verified, matrix operand order checked by reading, not run on hardware
//
#include <hip/hip_runtime.h>
#include <hip/hip_bf16.h>
#include <math.h>

// MI455X / gfx1250, wave32. Heavy contractions via v_wmma_f32_16x16x32_f16,
// weight panels staged to LDS via TDM (tensor_load_to_lds + s_wait_tensorcnt).

typedef __attribute__((ext_vector_type(16))) _Float16 v16h;
typedef __attribute__((ext_vector_type(8)))  _Float16 v8h;
typedef __attribute__((ext_vector_type(8)))  float    v8f;
typedef unsigned int u32x4 __attribute__((ext_vector_type(4)));
typedef unsigned int u32x8 __attribute__((ext_vector_type(8)));

#define BB    4
#define DIM   192
#define HH    64
#define WWI   64
#define NN    4096
#define HEADS 8
#define HD    24
#define LLW   9
#define PLEN  256

// ---------------- fragment helpers (ISA 7.12.2 layouts) ----------------
static __device__ __forceinline__ v16h pack16(v8h lo, v8h hi) {
  v16h r;
#pragma unroll
  for (int i = 0; i < 8; ++i) { r[i] = lo[i]; r[i + 8] = hi[i]; }
  return r;
}

// A (16x32): lane&15 = row; lanes<16 hold K {0..7,16..23}, lanes>=16 hold K {8..15,24..31}.
// Row-major f16, full K valid -> two aligned 128-bit loads.
static __device__ __forceinline__ v16h frag_a_f16(const _Float16* A, int ld, int m0, int k0) {
  int lane = threadIdx.x & 31;
  const _Float16* rp = A + (size_t)(m0 + (lane & 15)) * ld + k0 + ((lane >> 4) << 3);
  return pack16(*(const v8h*)rp, *(const v8h*)(rp + 16));
}

// Same pattern but K valid only to 24 (HD): upper half-lane group's second vector is zero.
static __device__ __forceinline__ v16h frag_f16_hd24(const _Float16* A, int ld, int m0) {
  int lane = threadIdx.x & 31;
  const _Float16* rp = A + (size_t)(m0 + (lane & 15)) * ld + ((lane >> 4) << 3);
  v8h lo = *(const v8h*)rp;
  v8h hi = {};
  if (lane < 16) hi = *(const v8h*)(rp + 16);   // K=16..23 valid, K=24..31 zero
  return pack16(lo, hi);
}

// B (32x16) from LDS weight panel: B[k][n] = Wl[n*192 + k] -> ds_load_b128 x2
static __device__ __forceinline__ v16h frag_b_lds(const _Float16* Wl, int k0) {
  int lane = threadIdx.x & 31;
  const _Float16* rp = Wl + (lane & 15) * DIM + k0 + ((lane >> 4) << 3);
  return pack16(*(const v8h*)rp, *(const v8h*)(rp + 16));
}

// B (32x16) K-major gather: B[k][n] = Mk[(k0+k)*ld + n0+n]; cols >= nvalid zero.
static __device__ __forceinline__ v16h frag_b_rm_f16(const _Float16* Mk, int ld, int k0, int n0, int nvalid) {
  int lane = threadIdx.x & 31;
  int n = n0 + (lane & 15);
  bool ok = (n < nvalid);
  int kb = (lane >> 4) << 3;
  v16h b;
#pragma unroll
  for (int i = 0; i < 8; ++i) {
    int k1 = k0 + kb + i, k2 = k0 + 16 + kb + i;
    b[i]     = ok ? Mk[(size_t)k1 * ld + n] : (_Float16)0.0f;
    b[i + 8] = ok ? Mk[(size_t)k2 * ld + n] : (_Float16)0.0f;
  }
  return b;
}

// ---------------- conversion / layout kernels ----------------
__global__ void cvt_f16(const float* __restrict__ s, _Float16* __restrict__ d, int n) {
  int i = blockIdx.x * blockDim.x + threadIdx.x;
  if (i < n) d[i] = (_Float16)s[i];
}

// x (B,C,H,W) f32 -> xt (B,N,C) f16
__global__ void cvt_x_nhwc(const float* __restrict__ x, _Float16* __restrict__ xt) {
  int i = blockIdx.x * blockDim.x + threadIdx.x;
  if (i >= BB * NN * DIM) return;
  int c = i % DIM;
  int n = (i / DIM) % NN;
  int b = i / (DIM * NN);
  xt[i] = (_Float16)x[((size_t)b * DIM + c) * NN + n];
}

// pe_w (O,C,3,3) f32 -> wt (O,tap,C) f16
__global__ void cvt_pew(const float* __restrict__ w, _Float16* __restrict__ wt) {
  int i = blockIdx.x * blockDim.x + threadIdx.x;
  if (i >= DIM * 9 * DIM) return;
  int c = i % DIM;
  int tap = (i / DIM) % 9;
  int o = i / (9 * DIM);
  wt[i] = (_Float16)w[((size_t)o * DIM + c) * 9 + tap];
}

// ---------------- conv3x3 as implicit GEMM (xt f16 NHWC, wt f16 (O,tap,C)) ----------------
__global__ __launch_bounds__(32) void conv3x3_wmma(const _Float16* __restrict__ xt,
                                                   const _Float16* __restrict__ wt,
                                                   const float* __restrict__ bias,
                                                   float* __restrict__ out) {
  int n0 = blockIdx.x << 4;
  int o0 = blockIdx.y << 4;
  int b  = blockIdx.z;
  int lane = threadIdx.x & 31;
  int m = n0 + (lane & 15);
  int y = m >> 6, xw = m & 63;
  int kb = (lane >> 4) << 3;
  int olane = o0 + (lane & 15);
  v8f acc = {};
  for (int tap = 0; tap < 9; ++tap) {
    int kh = tap / 3, kw = tap % 3;
    int ys = y + kh - 1, xs = xw + kw - 1;
    bool inb = (ys >= 0) && (ys < HH) && (xs >= 0) && (xs < WWI);
    int pix = inb ? (ys * WWI + xs) : 0;
    const _Float16* ap = xt + ((size_t)b * NN + pix) * DIM + kb;
    const _Float16* wp = wt + ((size_t)olane * 9 + tap) * DIM + kb;
    __builtin_prefetch(wp + DIM, 0, 3);
#pragma unroll
    for (int c0 = 0; c0 < DIM; c0 += 32) {
      v8h lo = {}, hi = {};
      if (inb) { lo = *(const v8h*)(ap + c0); hi = *(const v8h*)(ap + c0 + 16); }
      v16h a = pack16(lo, hi);
      v16h bf = pack16(*(const v8h*)(wp + c0), *(const v8h*)(wp + c0 + 16));
      acc = __builtin_amdgcn_wmma_f32_16x16x32_f16(false, a, false, bf, (short)0, acc, false, false);
    }
  }
  int col = lane & 15;
  int rb = (lane < 16) ? 0 : 8;
#pragma unroll
  for (int r = 0; r < 8; ++r) {
    int row = n0 + r + rb;
    out[((size_t)b * NN + row) * DIM + o0 + col] = acc[r] + bias[o0 + col];
  }
}

// ---------------- generic K=192 GEMM with TDM-staged weight panel ----------------
// C[m][o] = act(sum_k A[m][k] * W[o][k] + bias[o]);  8 waves/block, 128 rows/block.
__global__ __launch_bounds__(256) void gemm192_wmma(const _Float16* __restrict__ A,
                                                    const _Float16* __restrict__ Wh,
                                                    const float* __restrict__ bias,
                                                    float* __restrict__ C,
                                                    int Out, int act) {
  __shared__ _Float16 Wlds[16 * DIM];
  int o0 = blockIdx.y << 4;
  int wave = threadIdx.x >> 5;
  int m0 = (blockIdx.x << 7) + (wave << 4);

  if (wave == 0) {
    // TDM descriptor (CDNA5 ISA):
    // 2D tile 192x16 of 2-byte elements, tensor (192 x Out), stride 192.
    unsigned lds_off = (unsigned)(size_t)&Wlds[0];
    unsigned long long ga = (unsigned long long)(const void*)(Wh + (size_t)o0 * DIM);
    u32x4 g0;
    g0[0] = 1u;                                               // count=1, user mode
    g0[1] = lds_off;                                          // lds_addr
    g0[2] = (unsigned)ga;                                     // global_addr[31:0]
    g0[3] = ((unsigned)(ga >> 32) & 0x01FFFFFFu) | (2u << 30);// global_addr[56:32] | type=2
    u32x8 g1;
    g1[0] = 1u << 16;                                         // data_size = 2 bytes
    g1[1] = (unsigned)DIM << 16;                              // tensor_dim0[15:0]
    g1[2] = ((unsigned)Out & 0xFFFFu) << 16;                  // dim0 hi=0 | tensor_dim1 lo
    g1[3] = ((unsigned)Out >> 16) | ((unsigned)DIM << 16);    // tensor_dim1 hi | tile_dim0
    g1[4] = 16u;                                              // tile_dim1 = 16, tile_dim2 = 0
    g1[5] = (unsigned)DIM;                                    // tensor_dim0_stride lo32
    g1[6] = 0u;
    g1[7] = 0u;
    asm volatile("tensor_load_to_lds %0, %1" : : "s"(g0), "s"(g1) : "memory");
    __builtin_amdgcn_s_wait_tensorcnt(0);
  }
  __syncthreads();

  v8f acc = {};
#pragma unroll
  for (int k0 = 0; k0 < DIM; k0 += 32) {
    v16h a = frag_a_f16(A, DIM, m0, k0);
    v16h b = frag_b_lds(Wlds, k0);
    acc = __builtin_amdgcn_wmma_f32_16x16x32_f16(false, a, false, b, (short)0, acc, false, false);
  }
  int lane = threadIdx.x & 31;
  int col = lane & 15;
  int rb = (lane < 16) ? 0 : 8;
#pragma unroll
  for (int r = 0; r < 8; ++r) {
    int row = m0 + r + rb;
    int o = o0 + col;
    float v = acc[r] + bias[o];
    if (act == 1) v = 0.5f * v * (1.0f + erff(v * 0.70710678118654752f));  // exact GELU
    C[(size_t)row * Out + o] = v;
  }
}

// ---------------- row LayerNorm variants ----------------
__global__ void layernorm_f16out(const float* __restrict__ src, _Float16* __restrict__ dst,
                                 const float* __restrict__ g, const float* __restrict__ bt,
                                 int rows) {
  int row = (blockIdx.x << 3) + (threadIdx.x >> 5);
  if (row >= rows) return;
  int lane = threadIdx.x & 31;
  const float* rp = src + (size_t)row * DIM;
  float s = 0.f;
  for (int c = lane; c < DIM; c += 32) s += rp[c];
#pragma unroll
  for (int off = 16; off > 0; off >>= 1) s += __shfl_xor(s, off, 32);
  float mean = s / (float)DIM;
  float var = 0.f;
  for (int c = lane; c < DIM; c += 32) { float d = rp[c] - mean; var += d * d; }
#pragma unroll
  for (int off = 16; off > 0; off >>= 1) var += __shfl_xor(var, off, 32);
  float inv = rsqrtf(var / (float)DIM + 1e-5f);
  _Float16* op = dst + (size_t)row * DIM;
  for (int c = lane; c < DIM; c += 32) op[c] = (_Float16)((rp[c] - mean) * inv * g[c] + bt[c]);
}

__global__ void layernorm_transposed(const float* __restrict__ src, float* __restrict__ dst,
                                     const float* __restrict__ g, const float* __restrict__ bt,
                                     int rows) {
  int row = (blockIdx.x << 3) + (threadIdx.x >> 5);
  if (row >= rows) return;
  int lane = threadIdx.x & 31;
  const float* rp = src + (size_t)row * DIM;
  float s = 0.f;
  for (int c = lane; c < DIM; c += 32) s += rp[c];
#pragma unroll
  for (int off = 16; off > 0; off >>= 1) s += __shfl_xor(s, off, 32);
  float mean = s / (float)DIM;
  float var = 0.f;
  for (int c = lane; c < DIM; c += 32) { float d = rp[c] - mean; var += d * d; }
#pragma unroll
  for (int off = 16; off > 0; off >>= 1) var += __shfl_xor(var, off, 32);
  float inv = rsqrtf(var / (float)DIM + 1e-5f);
  int b = row / NN, n = row % NN;
  float* op = dst + (size_t)b * DIM * NN + n;
  for (int c = lane; c < DIM; c += 32) op[(size_t)c * NN] = (rp[c] - mean) * inv * g[c] + bt[c];
}

// ---------------- per-head L2 norm in place ----------------
__global__ void l2norm_heads(float* __restrict__ ptr, int rows, int ldr) {
  int idx = blockIdx.x * blockDim.x + threadIdx.x;
  if (idx >= rows * HEADS) return;
  int row = idx / HEADS, h = idx % HEADS;
  float* p = ptr + (size_t)row * ldr + h * HD;
  float s = 0.f;
  for (int d = 0; d < HD; ++d) s += p[d] * p[d];
  float inv = 1.0f / fmaxf(sqrtf(s), 1e-12f);
  for (int d = 0; d < HD; ++d) p[d] *= inv;
}

// ---------------- q: l2norm + query_embedding + softplus(temp) * seq_scale ----------------
__global__ void q_transform_k(const float* __restrict__ q, float* __restrict__ qn,
                              _Float16* __restrict__ qsh, const float* __restrict__ qe,
                              const float* __restrict__ temp, const float* __restrict__ seq) {
  int idx = blockIdx.x * blockDim.x + threadIdx.x;
  if (idx >= BB * NN * HEADS) return;
  int h = idx % HEADS;
  int row = idx / HEADS;
  int n = row % NN;
  const float* p = q + (size_t)row * DIM + h * HD;
  float s = 0.f;
  for (int d = 0; d < HD; ++d) s += p[d] * p[d];
  float inv = 1.0f / fmaxf(sqrtf(s), 1e-12f);
  float t = temp[h];
  float sp = (t > 20.f) ? t : log1pf(__expf(t));
  float sc = sp * seq[n];
  float* qnp = qn + (size_t)row * DIM + h * HD;
  _Float16* qsp = qsh + (size_t)row * DIM + h * HD;
  for (int d = 0; d < HD; ++d) {
    float v = p[d] * inv;
    qnp[d] = v;
    qsp[d] = (_Float16)((v + qe[h * HD + d]) * sc);
  }
}

// ---------------- 4x4 mean pool: sr_out (B,N,DIM) -> pooled (B,PLEN,DIM) ----------------
__global__ void pool4x4(const float* __restrict__ sr, float* __restrict__ pooled) {
  int idx = blockIdx.x * blockDim.x + threadIdx.x;
  if (idx >= BB * PLEN * DIM) return;
  int c = idx % DIM;
  int m = (idx / DIM) % PLEN;
  int b = idx / (DIM * PLEN);
  int ph = m >> 4, pw = m & 15;
  float s = 0.f;
#pragma unroll
  for (int i = 0; i < 4; ++i)
#pragma unroll
    for (int j = 0; j < 4; ++j) {
      int n = (ph * 4 + i) * WWI + (pw * 4 + j);
      s += sr[((size_t)b * NN + n) * DIM + c];
    }
  pooled[idx] = s * (1.0f / 16.0f);
}

// ---------------- CPB MLP: relu(table @ W1^T + b1) @ W2^T + b2 -> cpb[h][u] ----------------
__global__ void cpb_mlp(const float* __restrict__ table, const float* __restrict__ w1,
                        const float* __restrict__ b1, const float* __restrict__ w2,
                        const float* __restrict__ b2, float* __restrict__ out, int U) {
  __shared__ float hid[512];
  int u = blockIdx.x;
  if (u >= U) return;
  float t0 = table[(size_t)u * 2], t1 = table[(size_t)u * 2 + 1];
  for (int j = threadIdx.x; j < 512; j += blockDim.x)
    hid[j] = fmaxf(w1[j * 2] * t0 + w1[j * 2 + 1] * t1 + b1[j], 0.0f);
  __syncthreads();
  if (threadIdx.x < HEADS) {
    int h = threadIdx.x;
    float s = b2[h];
    for (int j = 0; j < 512; ++j) s += w2[h * 512 + j] * hid[j];
    out[(size_t)h * U + u] = s;
  }
}

// ---------------- fused foveal attention: 1 wave per (16 queries, head, batch) ----------------
#define STR 267  // 9 local + 256 pool logits, padded
__global__ __launch_bounds__(32) void foveal_attention(
    const _Float16* __restrict__ qsh, const float* __restrict__ qn,
    const float* __restrict__ kv, const _Float16* __restrict__ kvph,
    const float* __restrict__ cpb, const int* __restrict__ rel_idx,
    const float* __restrict__ rpb, const float* __restrict__ ltok,
    const float* __restrict__ lbias, const unsigned char* __restrict__ pad_mask,
    _Float16* __restrict__ atth, int U) {
  __shared__ float logit[16 * STR];
  __shared__ float wl[16 * LLW];
  __shared__ float xl[16 * HD];
  int n0 = blockIdx.x << 4;
  int h  = blockIdx.y;
  int b  = blockIdx.z;
  int lane = threadIdx.x & 31;
  const size_t rowbase = (size_t)b * NN;
  int col = lane & 15;
  int rb = (lane < 16) ? 0 : 8;

  // pool logits: q_scaled (16x24) @ k_pool^T (24x256) -> 16 WMMAs
  v16h aq = frag_f16_hd24(qsh + rowbase * DIM + h * HD, DIM, n0);
  const _Float16* kpool = kvph + (size_t)b * PLEN * (2 * DIM) + h * HD;
  for (int mt = 0; mt < PLEN; mt += 16) {
    v16h bk = frag_f16_hd24(kpool, 2 * DIM, mt);
    v8f d = {};
    d = __builtin_amdgcn_wmma_f32_16x16x32_f16(false, aq, false, bk, (short)0, d, false, false);
#pragma unroll
    for (int r = 0; r < 8; ++r) {
      int row = r + rb;
      int n = n0 + row;
      int m = mt + col;
      float bias = cpb[(size_t)h * U + rel_idx[(size_t)n * PLEN + m]];
      logit[row * STR + LLW + m] = d[r] + bias;
    }
  }

  // local 3x3 logits (masked with -inf at borders)
  for (int idx = lane; idx < 16 * LLW; idx += 32) {
    int row = idx / LLW, l = idx % LLW;
    int n = n0 + row;
    float v;
    if (pad_mask[(size_t)n * LLW + l]) {
      v = -INFINITY;
    } else {
      int dh = l / 3 - 1, dw = l % 3 - 1;
      int y = n >> 6, x = n & 63;
      int np = (y + dh) * WWI + (x + dw);
      const float* kp2 = kv + (rowbase + np) * (2 * DIM) + h * HD;
      const _Float16* qp = qsh + (rowbase + n) * DIM + h * HD;
      float s = rpb[h * LLW + l];
      for (int d2 = 0; d2 < HD; ++d2) s += (float)qp[d2] * kp2[d2];
      v = s;
    }
    logit[row * STR + l] = v;
  }
  __syncthreads();

  // softmax over 265 per row
  if (lane < 16) {
    float* lp = &logit[lane * STR];
    float mx = -INFINITY;
    for (int j = 0; j < LLW + PLEN; ++j) mx = fmaxf(mx, lp[j]);
    float sum = 0.f;
    for (int j = 0; j < LLW + PLEN; ++j) { float e = __expf(lp[j] - mx); lp[j] = e; sum += e; }
    float invs = 1.0f / sum;
    for (int j = 0; j < LLW + PLEN; ++j) lp[j] *= invs;
  }
  __syncthreads();

  // w_loc = qn @ learnable_tokens + learnable_bias + a_loc
  for (int idx = lane; idx < 16 * LLW; idx += 32) {
    int row = idx / LLW, l = idx % LLW;
    int n = n0 + row;
    const float* qnp = qn + (rowbase + n) * DIM + h * HD;
    float s = lbias[h * LLW + l] + logit[row * STR + l];
    for (int d2 = 0; d2 < HD; ++d2) s += qnp[d2] * ltok[((size_t)h * HD + d2) * LLW + l];
    wl[row * LLW + l] = s;
  }
  __syncthreads();

  // x_local = sum_l w_loc * v_local (v == 0 outside image)
  for (int idx = lane; idx < 16 * HD; idx += 32) {
    int row = idx / HD, d2 = idx % HD;
    int n = n0 + row;
    int y = n >> 6, x = n & 63;
    float s = 0.f;
#pragma unroll
    for (int l = 0; l < LLW; ++l) {
      int yy = y + l / 3 - 1, xx = x + l % 3 - 1;
      if (yy >= 0 && yy < HH && xx >= 0 && xx < WWI)
        s += wl[row * LLW + l] * kv[(rowbase + yy * WWI + xx) * (2 * DIM) + DIM + h * HD + d2];
    }
    xl[row * HD + d2] = s;
  }
  __syncthreads();

  // x_pool = a_pool (16x256) @ v_pool (256x24) -> 16 WMMAs (two 16-wide N tiles)
  const _Float16* vpool = kvph + (size_t)b * PLEN * (2 * DIM) + DIM + h * HD;
  v8f c0 = {}, c1 = {};
  for (int kt = 0; kt < PLEN; kt += 32) {
    int kb = (lane >> 4) << 3;
    int arow = lane & 15;
    v16h ap;
#pragma unroll
    for (int i = 0; i < 8; ++i) {
      ap[i]     = (_Float16)logit[arow * STR + LLW + kt + kb + i];
      ap[i + 8] = (_Float16)logit[arow * STR + LLW + kt + 16 + kb + i];
    }
    v16h b0 = frag_b_rm_f16(vpool, 2 * DIM, kt, 0, HD);
    c0 = __builtin_amdgcn_wmma_f32_16x16x32_f16(false, ap, false, b0, (short)0, c0, false, false);
    v16h b1 = frag_b_rm_f16(vpool, 2 * DIM, kt, 16, HD);
    c1 = __builtin_amdgcn_wmma_f32_16x16x32_f16(false, ap, false, b1, (short)0, c1, false, false);
  }
#pragma unroll
  for (int r = 0; r < 8; ++r) {
    int row = r + rb;
    int n = n0 + row;
    _Float16* op = atth + (rowbase + n) * DIM + h * HD;
    op[col] = (_Float16)(c0[r] + xl[row * HD + col]);
    if (col < HD - 16) op[16 + col] = (_Float16)(c1[r] + xl[row * HD + 16 + col]);
  }
}

// ---------------- host-side orchestration ----------------
extern "C" void kernel_launch(void* const* d_in, const int* in_sizes, int n_in,
                              void* d_out, int out_size, void* d_ws, size_t ws_size,
                              hipStream_t stream) {
  const float* x       = (const float*)d_in[0];
  const float* pe_w    = (const float*)d_in[1];
  const float* pe_b    = (const float*)d_in[2];
  const float* pe_ln_g = (const float*)d_in[3];
  const float* pe_ln_b = (const float*)d_in[4];
  const float* q_w     = (const float*)d_in[5];
  const float* q_b     = (const float*)d_in[6];
  const float* kv_w    = (const float*)d_in[7];
  const float* kv_b    = (const float*)d_in[8];
  const float* sr_w    = (const float*)d_in[9];
  const float* sr_b    = (const float*)d_in[10];
  const float* norm_g  = (const float*)d_in[11];
  const float* norm_b  = (const float*)d_in[12];
  const float* cpb1_w  = (const float*)d_in[13];
  const float* cpb1_b  = (const float*)d_in[14];
  const float* cpb2_w  = (const float*)d_in[15];
  const float* cpb2_b  = (const float*)d_in[16];
  const float* rpb     = (const float*)d_in[17];
  const float* ltok    = (const float*)d_in[18];
  const float* lbias   = (const float*)d_in[19];
  const float* qe      = (const float*)d_in[20];
  const float* temp    = (const float*)d_in[21];
  const float* proj_w  = (const float*)d_in[22];
  const float* proj_b  = (const float*)d_in[23];
  const float* o_ln_g  = (const float*)d_in[24];
  const float* o_ln_b  = (const float*)d_in[25];
  const int*   rel_idx = (const int*)d_in[26];
  const float* table   = (const float*)d_in[27];
  const float* seq     = (const float*)d_in[28];
  const unsigned char* pad_mask = (const unsigned char*)d_in[29];
  float* out = (float*)d_out;

  const int U = in_sizes[27] / 2;      // unique relative coords
  const size_t MN = (size_t)BB * NN;   // 16384 tokens

  char* base = (char*)d_ws;
  auto allocf = [&](size_t n) { float* r = (float*)base; base += ((n * 4 + 255) & ~(size_t)255); return r; };
  auto alloch = [&](size_t n) { _Float16* r = (_Float16*)base; base += ((n * 2 + 255) & ~(size_t)255); return r; };

  _Float16* xt      = alloch(MN * DIM + 64);
  _Float16* wt      = alloch((size_t)DIM * 9 * DIM + 64);
  _Float16* q_wh    = alloch((size_t)DIM * DIM + 64);
  _Float16* kv_wh   = alloch((size_t)2 * DIM * DIM + 64);
  _Float16* sr_wh   = alloch((size_t)DIM * DIM + 64);
  _Float16* proj_wh = alloch((size_t)DIM * DIM + 64);
  float*    tpre    = allocf(MN * DIM);
  _Float16* th      = alloch(MN * DIM + 64);
  float*    qbuf    = allocf(MN * DIM);
  float*    qnb     = allocf(MN * DIM);
  _Float16* qsh     = alloch(MN * DIM + 64);
  float*    kvbuf   = allocf(MN * 2 * DIM);
  float*    srbuf   = allocf(MN * DIM);
  float*    pooled  = allocf((size_t)BB * PLEN * DIM);
  _Float16* pooledh = alloch((size_t)BB * PLEN * DIM + 64);
  float*    kvpbuf  = allocf((size_t)BB * PLEN * 2 * DIM);
  _Float16* kvph    = alloch((size_t)BB * PLEN * 2 * DIM + 64);
  float*    cpbbuf  = allocf((size_t)HEADS * U);
  _Float16* atth    = alloch(MN * DIM + 64);
  float*    preout  = allocf(MN * DIM);

  // 0) f16 staging: input transpose, weights
  cvt_x_nhwc<<<(int)((MN * DIM + 255) / 256), 256, 0, stream>>>(x, xt);
  cvt_pew<<<(DIM * 9 * DIM + 255) / 256, 256, 0, stream>>>(pe_w, wt);
  cvt_f16<<<(DIM * DIM + 255) / 256, 256, 0, stream>>>(q_w, q_wh, DIM * DIM);
  cvt_f16<<<(2 * DIM * DIM + 255) / 256, 256, 0, stream>>>(kv_w, kv_wh, 2 * DIM * DIM);
  cvt_f16<<<(DIM * DIM + 255) / 256, 256, 0, stream>>>(sr_w, sr_wh, DIM * DIM);
  cvt_f16<<<(DIM * DIM + 255) / 256, 256, 0, stream>>>(proj_w, proj_wh, DIM * DIM);

  // 1) conv3x3 + bias -> tpre (B,N,DIM) f32
  conv3x3_wmma<<<dim3(NN / 16, DIM / 16, BB), 32, 0, stream>>>(xt, wt, pe_b, tpre);
  // 2) pe LayerNorm -> th (f16, GEMM operand)
  layernorm_f16out<<<(int)((MN + 7) / 8), 256, 0, stream>>>(tpre, th, pe_ln_g, pe_ln_b, (int)MN);
  // 3) q / kv / sr projections (TDM-staged weight panels)
  gemm192_wmma<<<dim3((int)(MN / 128), DIM / 16), 256, 0, stream>>>(th, q_wh, q_b, qbuf, DIM, 0);
  gemm192_wmma<<<dim3((int)(MN / 128), 2 * DIM / 16), 256, 0, stream>>>(th, kv_wh, kv_b, kvbuf, 2 * DIM, 0);
  gemm192_wmma<<<dim3((int)(MN / 128), DIM / 16), 256, 0, stream>>>(th, sr_wh, sr_b, srbuf, DIM, 1);
  // 4) l2norm k_loc, q transform
  l2norm_heads<<<(int)((MN * HEADS + 255) / 256), 256, 0, stream>>>(kvbuf, (int)MN, 2 * DIM);
  q_transform_k<<<(int)((MN * HEADS + 255) / 256), 256, 0, stream>>>(qbuf, qnb, qsh, qe, temp, seq);
  // 5) pool + LN (f16 out) + kv_p + l2norm k_pool + f16 copy
  pool4x4<<<(int)(((size_t)BB * PLEN * DIM + 255) / 256), 256, 0, stream>>>(srbuf, pooled);
  layernorm_f16out<<<(BB * PLEN + 7) / 8, 256, 0, stream>>>(pooled, pooledh, norm_g, norm_b, BB * PLEN);
  gemm192_wmma<<<dim3(BB * PLEN / 128, 2 * DIM / 16), 256, 0, stream>>>(pooledh, kv_wh, kv_b, kvpbuf, 2 * DIM, 0);
  l2norm_heads<<<(BB * PLEN * HEADS + 255) / 256, 256, 0, stream>>>(kvpbuf, BB * PLEN, 2 * DIM);
  cvt_f16<<<(int)(((size_t)BB * PLEN * 2 * DIM + 255) / 256), 256, 0, stream>>>(kvpbuf, kvph, BB * PLEN * 2 * DIM);
  // 6) CPB bias table
  cpb_mlp<<<U, 128, 0, stream>>>(table, cpb1_w, cpb1_b, cpb2_w, cpb2_b, cpbbuf, U);
  // 7) fused attention -> atth (f16)
  foveal_attention<<<dim3(NN / 16, HEADS, BB), 32, 0, stream>>>(
      qsh, qnb, kvbuf, kvph, cpbbuf, rel_idx, rpb, ltok, lbias, pad_mask, atth, U);
  // 8) proj + final LN (transposed store to (B,DIM,H,W))
  gemm192_wmma<<<dim3((int)(MN / 128), DIM / 16), 256, 0, stream>>>(atth, proj_wh, proj_b, preout, DIM, 0);
  layernorm_transposed<<<(int)((MN + 7) / 8), 256, 0, stream>>>(preout, out, o_ln_g, o_ln_b, (int)MN);
}
